// CRFTagger_24859270709546
// MI455X (gfx1250) — compile-verified, matrix operand order checked by reading, not run on hardware
//
#include <hip/hip_runtime.h>
#include <hip/hip_bf16.h>
#include <stdint.h>

// ---------------------------------------------------------------------------
// CRF tagger for MI455X (gfx1250).
//   scores  : (16384, 1024) f32
//   weights : (1024, 1024)  f32
//   tags    : (16384,)      i32
// out: 16384 predicted tags (as float) + 1 scalar (-logprob)
// ---------------------------------------------------------------------------

typedef float        v8f   __attribute__((ext_vector_type(8)));
typedef _Float16     v16h  __attribute__((ext_vector_type(16)));
typedef _Float16     v8h   __attribute__((ext_vector_type(8)));
typedef unsigned int u32x4 __attribute__((ext_vector_type(4)));
typedef int          i32x8 __attribute__((ext_vector_type(8)));
typedef int          i32x4 __attribute__((ext_vector_type(4)));

#define S_LEN 16384
#define NTAGS 1024
#define BEAM  64
#define CHUNK 512   // scan steps per precompute chunk (tile buffer = 12 MB)

#if defined(__HIP_DEVICE_COMPILE__) && __has_builtin(__builtin_amdgcn_tensor_load_to_lds) && __has_builtin(__builtin_amdgcn_s_wait_tensorcnt)
#define USE_TDM 1
#else
#define USE_TDM 0
#endif

#if USE_TDM
// 1-D TDM copy: nelem elements of (1<<dsz_code) bytes, global -> LDS.
// D# per CDNA5 ISA:
//   group0: count=1 | lds_addr | global_addr[56:0] | type=2
//   group1: data_size, tensor_dim0=nelem, tile_dim0=nelem, tile_dim1=0 (1-D),
//           tensor_dim0_stride=nelem
// clang-23 lane: 6-arg builtin (g0,u32x4)(g1,i32x8)(g2,i32x4)(g3,i32x4)(i32x8)(cpol)
__device__ __forceinline__ void tdm_load_1d(const void* gptr, void* lptr,
                                            int nelem, int dsz_code) {
  unsigned long long ga = (unsigned long long)(size_t)gptr;
  u32x4 g0;
  g0[0] = 1u;                                           // count=1, user mode
  g0[1] = (unsigned)(size_t)lptr;                       // lds_addr (low 32 of generic ptr)
  g0[2] = (unsigned)(ga & 0xffffffffu);                 // global_addr[31:0]
  g0[3] = (unsigned)((ga >> 32) & 0x01ffffffu) | (2u << 30); // ga[56:32] | type=2
  i32x8 g1;
  g1[0] = dsz_code << 16;                               // data_size (0=1B,1=2B,2=4B)
  g1[1] = (int)((unsigned)(nelem & 0xffff) << 16);      // tensor_dim0[15:0]
  g1[2] = (int)((unsigned)nelem >> 16);                 // tensor_dim0[31:16]
  g1[3] = (int)((unsigned)(nelem & 0xffff) << 16);      // tile_dim0
  g1[4] = 0;                                            // tile_dim1=0 (1-D), tile_dim2=0
  g1[5] = nelem;                                        // tensor_dim0_stride[31:0]
  g1[6] = 0;
  g1[7] = 0;
  i32x4 z4; z4[0] = 0; z4[1] = 0; z4[2] = 0; z4[3] = 0;
  i32x8 z8; z8[0] = 0; z8[1] = 0; z8[2] = 0; z8[3] = 0;
  z8[4] = 0; z8[5] = 0; z8[6] = 0; z8[7] = 0;
  __builtin_amdgcn_tensor_load_to_lds(g0, g1, z4, z4, z8, 0);
}
#endif

// ---------------------------------------------------------------------------
// Kernel 1: per-row top-64 via in-LDS bitonic sort (descending, index tiebreak)
// ---------------------------------------------------------------------------
__global__ void __launch_bounds__(256)
topk_kernel(const float* __restrict__ scores,
            float* __restrict__ bestS, int* __restrict__ bestT) {
  __shared__ float v[NTAGS];
  __shared__ int   ix[NTAGS];
  const int r = blockIdx.x, tid = threadIdx.x;
  for (int e = tid; e < NTAGS; e += 256) {
    v[e] = scores[(size_t)r * NTAGS + e];
    ix[e] = e;
  }
  __syncthreads();
  for (int k = 2; k <= NTAGS; k <<= 1) {
    for (int j = k >> 1; j > 0; j >>= 1) {
      for (int p = tid; p < NTAGS / 2; p += 256) {
        const int i = (p & (j - 1)) | ((p & ~(j - 1)) << 1);
        const int q = i | j;
        const bool desc = ((i & k) == 0);
        const float a = v[i], b = v[q];
        const bool sw = desc ? (a < b || (a == b && ix[i] > ix[q]))
                             : (a > b || (a == b && ix[i] < ix[q]));
        if (sw) {
          v[i] = b; v[q] = a;
          const int t0 = ix[i]; ix[i] = ix[q]; ix[q] = t0;
        }
      }
      __syncthreads();
    }
  }
  if (tid < BEAM) {
    bestS[(size_t)r * BEAM + tid] = v[tid];
    bestT[(size_t)r * BEAM + tid] = ix[tid];
  }
}

// ---------------------------------------------------------------------------
// Kernel 2: pre-gather per-step 64x64 W tiles (f32) and exp(W) tiles (f16)
// in WMMA-B operand order, for one chunk of scan steps. Fully parallel.
// E layout (per step, 4096 halfs):  idx = ((c*4+g)*32 + lane)*16 + m
//   lane in [0,32): column j = g*16 + (lane&15)
//   k = 32*c + kloc,  kloc = (m<8 ? m : m+8) + (lane>=16 ? 8 : 0)
// matching the 16-bit WMMA B VGPR striping.
// ---------------------------------------------------------------------------
__global__ void __launch_bounds__(256)
gather_kernel(const float* __restrict__ weights, const int* __restrict__ bestT,
              float* __restrict__ Wch, _Float16* __restrict__ Ech, int base) {
  const int s = base + blockIdx.x;
  __shared__ int pt[BEAM], ct[BEAM];
  const int tid = threadIdx.x;
  if (tid < BEAM)            pt[tid]      = bestT[(size_t)s * BEAM + tid];
  else if (tid < 2 * BEAM)   ct[tid - 64] = bestT[(size_t)(s + 1) * BEAM + (tid - 64)];
  __syncthreads();
  const size_t outb = (size_t)blockIdx.x * (BEAM * BEAM);
  for (int e = tid; e < BEAM * BEAM; e += 256) {
    const int i = e >> 6, j = e & 63;
    Wch[outb + e] = weights[(size_t)pt[i] * NTAGS + ct[j]];
  }
  for (int e = tid; e < BEAM * BEAM; e += 256) {
    const int c = e >> 11, g = (e >> 9) & 3, l = (e >> 4) & 31, m = e & 15;
    const int kloc = ((m < 8) ? m : (m + 8)) + ((l >= 16) ? 8 : 0);
    const int k = c * 32 + kloc;
    const int j = g * 16 + (l & 15);
    Ech[outb + e] = (_Float16)__expf(weights[(size_t)pt[k] * NTAGS + ct[j]]);
  }
}

// ---------------------------------------------------------------------------
// Kernel 3: sequential Viterbi + forward scan over one chunk. 1 block, 4 waves.
// Tiles streamed to LDS by the Tensor Data Mover (double buffered, TENSORcnt).
// Forward logsumexp inner sum via v_wmma_f32_16x16x32_f16 (wave 0).
// ---------------------------------------------------------------------------
__global__ void __launch_bounds__(128)
scan_kernel(const float* __restrict__ Wch, const _Float16* __restrict__ Ech,
            const float* __restrict__ bestS,
            float* __restrict__ vitG, float* __restrict__ fwdG,
            unsigned char* __restrict__ bps, int base, int len) {
  __shared__ alignas(64) float    Wt[2][BEAM * BEAM];   // 32 KB
  __shared__ alignas(64) _Float16 Et[2][BEAM * BEAM];   // 16 KB
  __shared__ float vit[BEAM], fwd[BEAM];
  __shared__ float pmax[4][BEAM];
  __shared__ int   parg[4][BEAM];
  __shared__ alignas(32) _Float16 aH[BEAM];
  __shared__ float rr[BEAM];
  __shared__ float Msh;

  const int tid  = threadIdx.x;
  const int lane = tid & 31;
  const int wv   = tid >> 5;

  if (tid < BEAM) {
    if (base == 0) { const float v = bestS[tid]; vit[tid] = v; fwd[tid] = v; }
    else           { vit[tid] = vitG[tid];       fwd[tid] = fwdG[tid]; }
  }

#if USE_TDM
  if (tid == 0) {                                   // prologue: tiles for s=0
    tdm_load_1d(Wch, &Wt[0][0], BEAM * BEAM, 2);
    tdm_load_1d(Ech, &Et[0][0], BEAM * BEAM, 1);
  }
#endif

  for (int s = 0; s < len; ++s) {
    const int buf = s & 1;
#if USE_TDM
    if (tid == 0 && s + 1 < len) {                  // prefetch next step
      tdm_load_1d(Wch + (size_t)(s + 1) * BEAM * BEAM, &Wt[buf ^ 1][0], BEAM * BEAM, 2);
      tdm_load_1d(Ech + (size_t)(s + 1) * BEAM * BEAM, &Et[buf ^ 1][0], BEAM * BEAM, 1);
    }
    if (s + 1 < len) __builtin_amdgcn_s_wait_tensorcnt(2);  // current tiles done
    else             __builtin_amdgcn_s_wait_tensorcnt(0);
#else
    __syncthreads();
    {
      const float*    gw = Wch + (size_t)s * BEAM * BEAM;
      const _Float16* ge = Ech + (size_t)s * BEAM * BEAM;
      for (int e = tid; e < BEAM * BEAM; e += 128) { Wt[buf][e] = gw[e]; Et[buf][e] = ge[e]; }
    }
#endif
    __syncthreads();  // B1: tiles + vit/fwd of this step visible

    const float* W = &Wt[buf][0];
    // ---- Viterbi partial max: wave wv covers i in [16wv,16wv+16), cols lane & lane+32
    {
      float m0 = -1e30f, m1 = -1e30f;
      int   b0 = 0,      b1 = 0;
      const int i0 = wv * 16;
#pragma unroll 4
      for (int q = 0; q < 16; ++q) {
        const int   i  = i0 + q;
        const float vi = vit[i];
        const float v0 = vi + W[i * BEAM + lane];
        const float v1 = vi + W[i * BEAM + lane + 32];
        if (v0 > m0) { m0 = v0; b0 = i; }
        if (v1 > m1) { m1 = v1; b1 = i; }
      }
      pmax[wv][lane] = m0;  pmax[wv][lane + 32] = m1;
      parg[wv][lane] = b0;  parg[wv][lane + 32] = b1;
    }
    // ---- wave 0: forward max M and a = exp(fwd - M) in f16
    if (wv == 0) {
      const float f0 = fwd[lane], f1 = fwd[lane + 32];
      float mm = fmaxf(f0, f1);
#pragma unroll
      for (int d = 16; d > 0; d >>= 1) mm = fmaxf(mm, __shfl_xor(mm, d, 32));
      aH[lane]      = (_Float16)__expf(f0 - mm);
      aH[lane + 32] = (_Float16)__expf(f1 - mm);
      if (lane == 0) Msh = mm;
    }
    __syncthreads();  // B2

    // ---- combine Viterbi partials, write new vit and backpointers
    if (tid < BEAM) {
      float best = pmax[0][tid];
      int   bb   = parg[0][tid];
#pragma unroll
      for (int p = 1; p < 4; ++p) {
        const float v = pmax[p][tid];
        if (v > best) { best = v; bb = parg[p][tid]; }
      }
      const float sj = bestS[(size_t)(base + s + 1) * BEAM + tid];
      vit[tid] = best + sj;
      bps[(size_t)(base + s) * BEAM + tid] = (unsigned char)bb;
      pmax[0][tid] = sj;  // stash cur_s for forward update
    }

    // ---- forward inner sum r = a(1x64) x exp(W)(64x64) via WMMA (wave 0)
#if __has_builtin(__builtin_amdgcn_wmma_f32_16x16x32_f16)
    if (wv == 0) {
      const _Float16* E = &Et[buf][0];
#pragma unroll
      for (int g = 0; g < 4; ++g) {                 // 16-column groups
        v8f acc = {};
#pragma unroll
        for (int c = 0; c < 2; ++c) {               // K chunks of 32
          const int ab = c * 32 + ((lane < 16) ? 0 : 8);
          const v8h alo = *(const v8h*)(aH + ab);        // k = ab + 0..7
          const v8h ahi = *(const v8h*)(aH + ab + 16);   // k = ab + 16..23
          v16h A;
#pragma unroll
          for (int m = 0; m < 8; ++m) { A[m] = alo[m]; A[m + 8] = ahi[m]; }
          const v16h B = *(const v16h*)(E + ((c * 4 + g) * 32 + lane) * 16);
          acc = __builtin_amdgcn_wmma_f32_16x16x32_f16(false, A, false, B,
                                                       (short)0, acc, false, false);
        }
        if (lane < 16) rr[g * 16 + lane] = acc[0];  // row M=0 of D
      }
    }
#else
    if (tid < BEAM) {  // scalar fallback decoding the E operand layout
      const _Float16* E = &Et[buf][0];
      float acc = 0.f;
      const int g = tid >> 4;
      for (int k = 0; k < BEAM; ++k) {
        const int c  = k >> 5, kl = k & 31;
        const int hi = (kl >> 3) & 1;
        const int m  = (kl & 16) ? (8 + (kl & 7)) : (kl & 7);
        const int l  = hi * 16 + (tid & 15);
        acc += (float)aH[k] * (float)E[((c * 4 + g) * 32 + l) * 16 + m];
      }
      rr[tid] = acc;
    }
#endif
    __syncthreads();  // B3

    if (tid < BEAM) fwd[tid] = pmax[0][tid] + Msh + __logf(rr[tid]);
  }

  __syncthreads();
  if (tid < BEAM) { vitG[tid] = vit[tid]; fwdG[tid] = fwd[tid]; }
}

// ---------------------------------------------------------------------------
// Kernel 4: backtrack (sequential pointer chase, bp array is L2-resident)
// ---------------------------------------------------------------------------
__global__ void backtrack_kernel(const float* __restrict__ vitF,
                                 const unsigned char* __restrict__ bps,
                                 int* __restrict__ idxs, int S) {
  if (threadIdx.x != 0) return;
  int last = 0;
  float bv = vitF[0];
  for (int i = 1; i < BEAM; ++i) {
    const float v = vitF[i];
    if (v > bv) { bv = v; last = i; }
  }
  idxs[S - 1] = last;
  int cur = last;
  for (int t = S - 2; t >= 0; --t) {
    cur = bps[(size_t)t * BEAM + cur];
    idxs[t] = cur;
  }
}

// ---------------------------------------------------------------------------
// Kernel 5: gather predicted tags
// ---------------------------------------------------------------------------
__global__ void __launch_bounds__(256)
finalize_kernel(const int* __restrict__ bestT, const int* __restrict__ idxs,
                float* __restrict__ out, int S) {
  const int t = blockIdx.x * blockDim.x + threadIdx.x;
  if (t < S) out[t] = (float)bestT[(size_t)t * BEAM + idxs[t]];
}

// ---------------------------------------------------------------------------
// Kernel 6/7: deterministic loss reduction:  -(base + crf - logZ)
// ---------------------------------------------------------------------------
__global__ void __launch_bounds__(256)
loss_partial(const float* __restrict__ scores, const float* __restrict__ weights,
             const int* __restrict__ tags, float* __restrict__ pb,
             float* __restrict__ pc, int S) {
  __shared__ float sb[256], sc[256];
  const int tid = threadIdx.x;
  const int gid = blockIdx.x * 256 + tid;
  const int stride = gridDim.x * 256;
  float b = 0.f, c = 0.f;
  for (int t = gid; t < S; t += stride) {
    b += scores[(size_t)t * NTAGS + tags[t]];
    if (t + 1 < S) c += weights[(size_t)tags[t] * NTAGS + tags[t + 1]];
  }
  sb[tid] = b; sc[tid] = c;
  __syncthreads();
  for (int d = 128; d > 0; d >>= 1) {
    if (tid < d) { sb[tid] += sb[tid + d]; sc[tid] += sc[tid + d]; }
    __syncthreads();
  }
  if (tid == 0) { pb[blockIdx.x] = sb[0]; pc[blockIdx.x] = sc[0]; }
}

__global__ void __launch_bounds__(64)
loss_final(const float* __restrict__ pb, const float* __restrict__ pc, int nparts,
           const float* __restrict__ fwdF, float* __restrict__ out, int S) {
  __shared__ float s1[64], s2[64], sf[64];
  const int tid = threadIdx.x;
  float b = 0.f, c = 0.f;
  for (int i = tid; i < nparts; i += 64) { b += pb[i]; c += pc[i]; }
  s1[tid] = b; s2[tid] = c; sf[tid] = fwdF[tid];
  __syncthreads();
  for (int d = 32; d > 0; d >>= 1) {
    if (tid < d) { s1[tid] += s1[tid + d]; s2[tid] += s2[tid + d]; }
    __syncthreads();
  }
  if (tid == 0) {
    float m = sf[0];
    for (int i = 1; i < BEAM; ++i) m = fmaxf(m, sf[i]);
    float sum = 0.f;
    for (int i = 0; i < BEAM; ++i) sum += __expf(sf[i] - m);
    const float logZ = m + __logf(sum);
    out[S] = -(s1[0] + s2[0] - logZ);
  }
}

// ---------------------------------------------------------------------------
// Host launcher
// ---------------------------------------------------------------------------
extern "C" void kernel_launch(void* const* d_in, const int* in_sizes, int n_in,
                              void* d_out, int out_size, void* d_ws, size_t ws_size,
                              hipStream_t stream) {
  (void)in_sizes; (void)n_in; (void)out_size; (void)ws_size;
  const float* scores  = (const float*)d_in[0];
  const float* weights = (const float*)d_in[1];
  const int*   tags    = (const int*)d_in[2];
  float* out = (float*)d_out;
  const int S = S_LEN;
  const int STEPS = S - 1;

  char* ws = (char*)d_ws;
  size_t off = 0;
  auto take = [&](size_t bytes) -> char* {
    char* p = ws + off;
    off = (off + bytes + 255) & ~(size_t)255;
    return p;
  };
  float*         bestS = (float*)take((size_t)S * BEAM * sizeof(float));   // 4 MB
  int*           bestT = (int*)  take((size_t)S * BEAM * sizeof(int));     // 4 MB
  unsigned char* bps   = (unsigned char*)take((size_t)STEPS * BEAM);       // 1 MB
  float*         vitF  = (float*)take(BEAM * sizeof(float));
  float*         fwdF  = (float*)take(BEAM * sizeof(float));
  int*           idxs  = (int*)  take((size_t)S * sizeof(int));            // 64 KB
  float*         pb    = (float*)take(64 * sizeof(float));
  float*         pc    = (float*)take(64 * sizeof(float));
  float*         Wch   = (float*)take((size_t)CHUNK * BEAM * BEAM * sizeof(float));    // 8 MB
  _Float16*      Ech   = (_Float16*)take((size_t)CHUNK * BEAM * BEAM * sizeof(_Float16)); // 4 MB

  topk_kernel<<<S, 256, 0, stream>>>(scores, bestS, bestT);

  const int nch = (STEPS + CHUNK - 1) / CHUNK;
  for (int c = 0; c < nch; ++c) {
    const int b0 = c * CHUNK;
    int len = STEPS - b0;
    if (len > CHUNK) len = CHUNK;
    gather_kernel<<<len, 256, 0, stream>>>(weights, bestT, Wch, Ech, b0);
    scan_kernel<<<1, 128, 0, stream>>>(Wch, Ech, bestS, vitF, fwdF, bps, b0, len);
  }

  backtrack_kernel<<<1, 32, 0, stream>>>(vitF, bps, idxs, S);
  finalize_kernel<<<S / 256, 256, 0, stream>>>(bestT, idxs, out, S);
  loss_partial<<<64, 256, 0, stream>>>(scores, weights, tags, pb, pc, S);
  loss_final<<<1, 64, 0, stream>>>(pb, pc, 64, fwdF, out, S);
}
